// Experts_68255620268123
// MI455X (gfx1250) — matrix operand order repaired and checked by reading, not compiled, not run on hardware
//
#include <hip/hip_runtime.h>

// Problem constants (match reference)
#define BB   4
#define EE   8
#define NN   1024
#define DDIM 1024
#define HDIM 4096

// Tiling
#define MBLK  32   // rows of N per workgroup
#define KC    64   // H-chunk width
#define SSTEP 4    // K-steps (of 32) staged per GEMM1 slab

typedef __bf16 bf16;
typedef __attribute__((ext_vector_type(4)))  bf16  bf16x4;
typedef __attribute__((ext_vector_type(8)))  bf16  bf16x8;
typedef __attribute__((ext_vector_type(16))) bf16  bf16x16;
typedef __attribute__((ext_vector_type(8)))  float v8f;
typedef __attribute__((ext_vector_type(4)))  float f32x4;

// LDS layout (bf16 element counts)
#define XROW 1032                         // 1024 + 8 pad
#define HROW 72                           // 64 + 8 pad
#define LDS_X_ELEMS     (MBLK * XROW)     // 33,024  (66 KB)
#define LDS_H_ELEMS     (MBLK * HROW)     //  2,304  (4.5 KB)
#define W1STEP_ELEMS    (4 * 32 * 16)     //  2,048 per K-step (B-lane order)
#define LDS_W1SLAB_ELEMS (SSTEP * W1STEP_ELEMS) // 8,192 (16 KB) per buffer, x2
#define LDS_W2_ELEMS    (64 * 2 * 32 * 16)// 65,536 (128 KB), aliased as f32 out-stage
#define SMEM_BYTES ((size_t)(LDS_X_ELEMS + LDS_H_ELEMS + 2*LDS_W1SLAB_ELEMS + LDS_W2_ELEMS) * 2)

static __device__ __forceinline__ bf16x16 cat8(bf16x8 lo, bf16x8 hi) {
  return __builtin_shufflevector(lo, hi, 0,1,2,3,4,5,6,7,8,9,10,11,12,13,14,15);
}

static __device__ __forceinline__ v8f wmma_bf16(bf16x16 a, bf16x16 b, v8f c) {
  // D = A*B + C, f32 accum -> v_wmma_f32_16x16x32_bf16
  return __builtin_amdgcn_wmma_f32_16x16x32_bf16(false, a, false, b, (short)0, c, false, false);
}

__global__ __launch_bounds__(256, 1)
void experts_fused_kernel(const float* __restrict__ x,
                          const float* __restrict__ w1,
                          const float* __restrict__ b1,
                          const float* __restrict__ w2,
                          const float* __restrict__ b2,
                          float* __restrict__ out) {
  extern __shared__ char smem[];
  bf16*  sx   = (bf16*)smem;                      // x block, row-major bf16 (A layout)
  bf16*  shl  = sx  + LDS_X_ELEMS;                // gelu(h) chunk, row-major bf16
  bf16*  sw1  = shl + LDS_H_ELEMS;                // w1 slabs, B-lane order (x2)
  bf16*  sw2  = sw1 + 2 * LDS_W1SLAB_ELEMS;       // w2 chunk, B-lane order
  float* sout = (float*)sw2;                      // aliased f32 output staging

  const int tid  = threadIdx.x;
  const int lane = tid & 31;
  const int w    = tid >> 5;      // wave 0..7
  const int r    = w & 1;         // GEMM1 row-tile (0..1)
  const int wp   = w >> 1;        // GEMM1 wave pair 0..3

  const int bb   = blockIdx.z;
  const int ee   = blockIdx.y;
  const int row0 = blockIdx.x * MBLK;

  const float* xg  = x  + ((size_t)(bb * EE + ee) * NN + row0) * DDIM;
  const float* w1g = w1 + (size_t)ee * DDIM * HDIM;
  const float* w2g = w2 + (size_t)ee * HDIM * DDIM;
  const float* b1g = b1 + (size_t)ee * HDIM;
  const float* b2g = b2 + (size_t)ee * DDIM;
  float*       og  = out + ((size_t)(bb * EE + ee) * NN + row0) * DDIM;

  // ---- Stage x block: f32 -> bf16, row-major (read-once -> non-temporal) --
  for (int i = 0; i < (MBLK * DDIM / 4) / 256; ++i) {
    int idx4 = i * 256 + tid;
    int row  = idx4 >> 8;
    int col  = (idx4 & 255) * 4;
    f32x4 v = __builtin_nontemporal_load((const f32x4*)(xg + (size_t)row * DDIM + col));
    bf16x4 p = { (bf16)v.x, (bf16)v.y, (bf16)v.z, (bf16)v.w };
    *(bf16x4*)(sx + row * XROW + col) = p;
  }

  // GEMM2 accumulators: wave w owns BOTH row halves x 8 col tiles:
  //   acc[j]   -> rows 0..15,  col tile w*8+j
  //   acc[8+j] -> rows 16..31, col tile w*8+j
  v8f acc[16];
#pragma unroll
  for (int j = 0; j < 16; ++j) acc[j] = (v8f){0,0,0,0,0,0,0,0};

  const int asel = (lane >> 4) * 16;         // A-operand K sub-select per lane half
  const int arow = r * 16 + (lane & 15);     // GEMM1 A-operand row for this lane
  const int blk  = (wp * 32 + lane) << 4;    // GEMM1 B-lane block offset

  for (int ko = 0; ko < HDIM / KC; ++ko) {
    const int h0 = ko * KC;
    __syncthreads();   // protect shl/sw2 from previous-iteration readers

    // ---- Stage w2 chunk [64h][1024d] -> B-lane-order bf16 blocks ----------
    // Read 4 floats along h (K dim) so the 4 bf16 are contiguous: 1 ds_store_b64.
    for (int i = 0; i < 64; ++i) {
      int hh4 = i >> 2;                       // 0..15 (h row group of 4)
      int d   = ((i & 3) << 8) + tid;         // 0..1023
      int ct  = d >> 4;
      int kk  = hh4 >> 3;
      int n   = (d & 15) | ((hh4 & 4) << 2);
      const float* gp = w2g + (size_t)(h0 + hh4 * 4) * DDIM + d;
      bf16x4 p = { (bf16)gp[0], (bf16)gp[DDIM], (bf16)gp[2 * DDIM], (bf16)gp[3 * DDIM] };
      *(bf16x4*)(sw2 + (((ct * 2 + kk) * 32 + n) << 4) + ((hh4 & 3) << 2)) = p;
    }

    // ---- Stage w1 slab 0 into buffer 0 ------------------------------------
    {
#pragma unroll
      for (int i = 0; i < 8; ++i) {
        int idx = i * 256 + tid;
        int dd4 = idx >> 6;                   // 0..31 (d row group of 4)
        int h   = idx & 63;
        int dd  = dd4 << 2;                   // 0..124
        int ss  = dd >> 5;                    // K-step within slab
        int ct  = h >> 4;
        int n   = (h & 15) | (dd & 16);
        const float* gp = w1g + (size_t)dd * HDIM + h0 + h;
        bf16x4 p = { (bf16)gp[0], (bf16)gp[HDIM], (bf16)gp[2 * (size_t)HDIM], (bf16)gp[3 * (size_t)HDIM] };
        *(bf16x4*)(sw1 + ss * W1STEP_ELEMS + ((ct * 32 + n) << 4) + ((dd4 & 3) << 2)) = p;
      }
    }
    __syncthreads();

    // ---- GEMM1: h_tile[r, wp] = x @ w1[:, h0:h0+64], slabs of 4 K-steps ---
    v8f acc1 = (v8f){0,0,0,0,0,0,0,0};
    const int NSLAB = DDIM / (32 * SSTEP);   // 8
    for (int slab = 0; slab < NSLAB; ++slab) {
      // software pipeline: stage next slab into the other buffer
      if (slab + 1 < NSLAB) {
        const int nb = (slab + 1) & 1;
        const float* w1s = w1g + (size_t)(slab + 1) * (SSTEP * 32) * HDIM;
#pragma unroll
        for (int i = 0; i < 8; ++i) {
          int idx = i * 256 + tid;
          int dd4 = idx >> 6;
          int h   = idx & 63;
          int dd  = dd4 << 2;
          int ss  = dd >> 5;
          int ct  = h >> 4;
          int n   = (h & 15) | (dd & 16);
          const float* gp = w1s + (size_t)dd * HDIM + h0 + h;
          bf16x4 p = { (bf16)gp[0], (bf16)gp[HDIM], (bf16)gp[2 * (size_t)HDIM], (bf16)gp[3 * (size_t)HDIM] };
          *(bf16x4*)(sw1 + nb * LDS_W1SLAB_ELEMS + ss * W1STEP_ELEMS + ((ct * 32 + n) << 4) + ((dd4 & 3) << 2)) = p;
        }
      }
      // load all operands for the slab first, then 4 back-to-back WMMAs
      const bf16* buf = sw1 + (slab & 1) * LDS_W1SLAB_ELEMS;
      bf16x16 av[SSTEP], bv[SSTEP];
#pragma unroll
      for (int ss = 0; ss < SSTEP; ++ss) {
        int s = slab * SSTEP + ss;
        const bf16* ap = sx + arow * XROW + s * 32 + asel;
        av[ss] = cat8(*(const bf16x8*)ap, *(const bf16x8*)(ap + 8));
        const bf16* bp = buf + ss * W1STEP_ELEMS + blk;
        bv[ss] = cat8(*(const bf16x8*)bp, *(const bf16x8*)(bp + 8));
      }
#pragma unroll
      for (int ss = 0; ss < SSTEP; ++ss) acc1 = wmma_bf16(av[ss], bv[ss], acc1);
      __syncthreads();
    }

    // ---- bias + exact GELU, store h tile to LDS (A layout row-major) ------
    {
      int hl = wp * 16 + (lane & 15);        // C layout: N = lane%16
      float bias = b1g[h0 + hl];
#pragma unroll
      for (int g = 0; g < 8; ++g) {
        int m = g + ((lane >> 4) << 3);      // C layout: M = vgpr (+8 hi lanes)
        float v  = acc1[g] + bias;
        float ge = 0.5f * v * (1.0f + erff(v * 0.70710678118654752f));
        shl[(r * 16 + m) * HROW + hl] = (bf16)ge;
      }
    }
    __syncthreads();

    // ---- GEMM2 accumulate: C += gelu(h)[32,64] @ w2[h0:h0+64, :] ----------
    // Each wave: both row halves, 8 col tiles; 1 B pair feeds 2 WMMAs.
    {
      const int arowL = (lane & 15) * HROW;          // rows 0..15
      const int arowH = (16 + (lane & 15)) * HROW;   // rows 16..31
#pragma unroll
      for (int kk = 0; kk < 2; ++kk) {
        const bf16* apL = shl + arowL + kk * 32 + asel;
        const bf16* apH = shl + arowH + kk * 32 + asel;
        bf16x16 a0 = cat8(*(const bf16x8*)apL, *(const bf16x8*)(apL + 8));
        bf16x16 a1 = cat8(*(const bf16x8*)apH, *(const bf16x8*)(apH + 8));

        // distance-2 rotating prefetch of B tiles
        const bf16* bp0 = sw2 + ((((w * 8 + 0) * 2 + kk) * 32 + lane) << 4);
        const bf16* bp1 = sw2 + ((((w * 8 + 1) * 2 + kk) * 32 + lane) << 4);
        bf16x16 bcur = cat8(*(const bf16x8*)bp0, *(const bf16x8*)(bp0 + 8));
        bf16x16 bnxt = cat8(*(const bf16x8*)bp1, *(const bf16x8*)(bp1 + 8));
#pragma unroll
        for (int j = 0; j < 8; ++j) {
          bf16x16 bfut = bcur;
          if (j + 2 < 8) {
            const bf16* bp = sw2 + ((((w * 8 + j + 2) * 2 + kk) * 32 + lane) << 4);
            bfut = cat8(*(const bf16x8*)bp, *(const bf16x8*)(bp + 8));
          }
          acc[j]     = wmma_bf16(a0, bcur, acc[j]);
          acc[8 + j] = wmma_bf16(a1, bcur, acc[8 + j]);
          bcur = bnxt;
          bnxt = bfut;
        }
      }
    }
  }

  // ---- Epilogue: bias2, stage via LDS (aliases sw2), coalesced NT store ---
  __syncthreads();   // all waves done reading sw2 before aliasing as sout
#pragma unroll
  for (int j = 0; j < 8; ++j) {
    int d = w * 128 + j * 16 + (lane & 15);
    float bias2 = b2g[d];
#pragma unroll
    for (int g = 0; g < 8; ++g) {
      int m = g + ((lane >> 4) << 3);
      sout[m * DDIM + d]        = acc[j][g]     + bias2;   // rows 0..15
      sout[(16 + m) * DDIM + d] = acc[8 + j][g] + bias2;   // rows 16..31
    }
  }
  __syncthreads();
  for (int i = 0; i < (MBLK * DDIM / 4) / 256; ++i) {
    int idx4 = i * 256 + tid;
    int row  = idx4 >> 8;
    int col  = (idx4 & 255) * 4;
    f32x4 v = *(const f32x4*)(sout + row * DDIM + col);
    __builtin_nontemporal_store(v, (f32x4*)(og + (size_t)row * DDIM + col));
  }
}

extern "C" void kernel_launch(void* const* d_in, const int* in_sizes, int n_in,
                              void* d_out, int out_size, void* d_ws, size_t ws_size,
                              hipStream_t stream) {
  (void)in_sizes; (void)n_in; (void)out_size; (void)d_ws; (void)ws_size;
  const float* x  = (const float*)d_in[0];
  const float* w1 = (const float*)d_in[1];
  const float* b1 = (const float*)d_in[2];
  const float* w2 = (const float*)d_in[3];
  const float* b2 = (const float*)d_in[4];
  float* out = (float*)d_out;

  dim3 grid(NN / MBLK, EE, BB);   // 32 x 8 x 4 = 1024 workgroups
  dim3 block(256);                // 8 waves (wave32)
  experts_fused_kernel<<<grid, block, SMEM_BYTES, stream>>>(x, w1, b1, w2, b2, out);
}
